// SE3Transformer_62483184222602
// MI455X (gfx1250) — compile-verified, matrix operand order checked by reading, not run on hardware
//
#include <hip/hip_runtime.h>
#include <math.h>

typedef __attribute__((ext_vector_type(16))) _Float16 v16h;
typedef __attribute__((ext_vector_type(8)))  _Float16 v8h;
typedef __attribute__((ext_vector_type(8)))  float    v8f;

#define NCH  16
#define MIDC 32
#define F0C  67
#define Y0F  0.28209479177387814f
#define C1F  0.4886025119029199f

static __device__ __forceinline__ v8f wmma16(v16h a, v16h b, v8f c) {
  // D = A(16x32 f16) * B(32x16 f16) + C(16x16 f32)
  return __builtin_amdgcn_wmma_f32_16x16x32_f16(false, a, false, b, (short)0, c,
                                                false, false);
}

// ds_swizzle with immediate pattern (group-of-32: offset = xor<<10|or<<5|and)
template <int IMM>
static __device__ __forceinline__ float swz(float x) {
  return __int_as_float(__builtin_amdgcn_ds_swizzle(__float_as_int(x), IMM));
}
// add value from partner lane (lane ^ 16)
static __device__ __forceinline__ float red_pair32(float x) {
  return x + swz<(16 << 10) | 0x1f>(x);
}

// h tile lives in LDS in *physical A-layout order*: row M, position
// p = (K&7) | ((K&8)<<1) | ((K&16)>>1)   (swap bits 3<->4 of K, involution).
// lane l reads its v16h contiguously:
//   lanes 0-15:  phys 0..15  = K {0..7,16..23}
//   lanes 16-31: phys 16..31 = K {8..15,24..31}
static __device__ __forceinline__ v16h load_A(const _Float16* hb, int l) {
  return *(const v16h*)&hb[(l & 15) * MIDC + (l & 16)];
}

// ---------------------------------------------------------------------------
__global__ void fill_kernel(float* __restrict__ p, int n, float v) {
  int i = blockIdx.x * blockDim.x + threadIdx.x;
  if (i < n) p[i] = v;
}

// ---------------------------------------------------------------------------
// node embedding: h0 = elu(nf@W1+b1)@W2+b2   (one thread per node, VALU)
__global__ __launch_bounds__(256) void embed_kernel(
    const float* __restrict__ nf, const float* __restrict__ W1,
    const float* __restrict__ b1, const float* __restrict__ W2,
    const float* __restrict__ b2, float* __restrict__ h0, int N) {
  int n = blockIdx.x * blockDim.x + threadIdx.x;
  if (n >= N) return;
  float row[F0C];
#pragma unroll
  for (int k = 0; k < F0C; ++k) row[k] = nf[n * F0C + k];
  float acc[NCH];
#pragma unroll
  for (int c = 0; c < NCH; ++c) acc[c] = b2[c];
  for (int j = 0; j < F0C; ++j) {
    float t = b1[j];
#pragma unroll
    for (int k = 0; k < F0C; ++k) t = fmaf(row[k], W1[k * F0C + j], t);
    t = (t > 0.f) ? t : (expf(t) - 1.f);  // elu
#pragma unroll
    for (int c = 0; c < NCH; ++c) acc[c] = fmaf(t, W2[j * NCH + c], acc[c]);
  }
#pragma unroll
  for (int c = 0; c < NCH; ++c) h0[n * NCH + c] = acc[c];
}

// ---------------------------------------------------------------------------
__global__ void deg_kernel(const int* __restrict__ dst, float* __restrict__ deg,
                           int E) {
  int i = blockIdx.x * blockDim.x + threadIdx.x;
  if (i < E) atomicAdd(&deg[dst[i]], 1.0f);
}

// ---------------------------------------------------------------------------
// One step of the fused L3 + einsum('eoi,ei->eo'): WMMA tile TAU yields
// R[e, o, i=TAU] in C-layout (zero C; rb3 bias handled by a dedicated
// bias WMMA that initializes acc); multiply by group broadcast of s[e,TAU]
// (ds_swizzle BCASTX16: and=0x10, or=TAU). Broadcasts shared by both nets.
#define TAU_STEP(TAU)                                                          \
  do {                                                                         \
    float bc[8];                                                               \
    bc[0] = swz<0x10 | ((TAU) << 5)>(Sr[0]);                                   \
    bc[1] = swz<0x10 | ((TAU) << 5)>(Sr[1]);                                   \
    bc[2] = swz<0x10 | ((TAU) << 5)>(Sr[2]);                                   \
    bc[3] = swz<0x10 | ((TAU) << 5)>(Sr[3]);                                   \
    bc[4] = swz<0x10 | ((TAU) << 5)>(Sr[4]);                                   \
    bc[5] = swz<0x10 | ((TAU) << 5)>(Sr[5]);                                   \
    bc[6] = swz<0x10 | ((TAU) << 5)>(Sr[6]);                                   \
    bc[7] = swz<0x10 | ((TAU) << 5)>(Sr[7]);                                   \
    {                                                                          \
      const v16h bt = *(const v16h*)&B3lds[(((TAU)) * 32 + l) * 16];           \
      const v8f d = wmma16(a2s[0], bt, zc);                                    \
      acc0[0] = fmaf(d[0], bc[0], acc0[0]);                                    \
      acc0[1] = fmaf(d[1], bc[1], acc0[1]);                                    \
      acc0[2] = fmaf(d[2], bc[2], acc0[2]);                                    \
      acc0[3] = fmaf(d[3], bc[3], acc0[3]);                                    \
      acc0[4] = fmaf(d[4], bc[4], acc0[4]);                                    \
      acc0[5] = fmaf(d[5], bc[5], acc0[5]);                                    \
      acc0[6] = fmaf(d[6], bc[6], acc0[6]);                                    \
      acc0[7] = fmaf(d[7], bc[7], acc0[7]);                                    \
    }                                                                          \
    if (NNETS == 2) {                                                          \
      const v16h bt = *(const v16h*)&B3lds[((16 + (TAU)) * 32 + l) * 16];      \
      const v8f d = wmma16(a2s[1], bt, zc);                                    \
      acc1[0] = fmaf(d[0], bc[0], acc1[0]);                                    \
      acc1[1] = fmaf(d[1], bc[1], acc1[1]);                                    \
      acc1[2] = fmaf(d[2], bc[2], acc1[2]);                                    \
      acc1[3] = fmaf(d[3], bc[3], acc1[3]);                                    \
      acc1[4] = fmaf(d[4], bc[4], acc1[4]);                                    \
      acc1[5] = fmaf(d[5], bc[5], acc1[5]);                                    \
      acc1[6] = fmaf(d[6], bc[6], acc1[6]);                                    \
      acc1[7] = fmaf(d[7], bc[7], acc1[7]);                                    \
    }                                                                          \
  } while (0)

// fused edge kernel: radial MLP (WMMA) + einsum contraction + scatter-add.
// One wave = one tile of 16 edges. 8 waves / block.
template <int NNETS>
__global__ __launch_bounds__(256) void edge_kernel(
    const float* __restrict__ pos, const float* __restrict__ edge_w,
    const int* __restrict__ src, const int* __restrict__ dst,
    const float* __restrict__ rW1, const float* __restrict__ rb1,
    const float* __restrict__ ln1g, const float* __restrict__ ln1b,
    const float* __restrict__ rW2, const float* __restrict__ rb2,
    const float* __restrict__ ln2g, const float* __restrict__ ln2b,
    const float* __restrict__ rW3, const float* __restrict__ rb3,
    const float* __restrict__ h0in, float* __restrict__ out0raw,
    float* __restrict__ out1raw, int net0, int net1, int nTiles) {
  __shared__ __align__(32) _Float16 B3lds[2 * 16 * 32 * 16];  // rW3 B-layout
  __shared__ __align__(32) _Float16 B2lds[2 * 2 * 32 * 16];   // rW2 B-layout
  __shared__ __align__(32) _Float16 Bblds[2 * 32 * 16];  // rb3 bias-GEMM B
  __shared__ __align__(16) float ln2gb[2 * 2 * 32];      // g/b in phys-K order
  __shared__ __align__(32) _Float16 hbuf[8][16 * MIDC];  // per-wave h tile
  __shared__ int dstStash[8 * 16];
  __shared__ float y1Stash[8 * 16 * 3];

  const int tid = threadIdx.x;
  // ---- preamble: convert & permute weights into WMMA layouts --------------
  for (int idx = tid; idx < NNETS * 16 * 32 * 16; idx += 256) {
    const int h = idx & 15, L = (idx >> 4) & 31, tau = (idx >> 9) & 15,
              ns = idx >> 13;
    const int net = (ns == 0) ? net0 : net1;
    const int m = ((h < 8) ? h : (h + 8)) + ((L >= 16) ? 8 : 0);  // K index
    const int o = L & 15;                                          // column
    B3lds[idx] = (_Float16)rW3[(net * MIDC + m) * 256 + o * 16 + tau];
  }
  for (int idx = tid; idx < NNETS * 2 * 32 * 16; idx += 256) {
    const int h = idx & 15, L = (idx >> 4) & 31, ct = (idx >> 9) & 1,
              ns = idx >> 10;
    const int net = (ns == 0) ? net0 : net1;
    const int m = ((h < 8) ? h : (h + 8)) + ((L >= 16) ? 8 : 0);
    B2lds[idx] = (_Float16)rW2[(net * MIDC + m) * MIDC + ct * 16 + (L & 15)];
  }
  for (int idx = tid; idx < NNETS * 32 * 16; idx += 256) {
    // bias GEMM B: Bb[k][o] = rb3[o*16+k] for k<16, 0 for k>=16
    const int h = idx & 15, L = (idx >> 4) & 31, ns = idx >> 9;
    const int net = (ns == 0) ? net0 : net1;
    const int K = ((h < 8) ? h : (h + 8)) + ((L >= 16) ? 8 : 0);
    Bblds[idx] =
        (K < 16) ? (_Float16)rb3[net * 256 + (L & 15) * 16 + K] : (_Float16)0.f;
  }
  for (int idx = tid; idx < NNETS * 2 * 32; idx += 256) {
    const int p = idx & 31, which = (idx >> 5) & 1, ns = idx >> 6;
    const int net = (ns == 0) ? net0 : net1;
    const int K = (p & 7) | ((p & 8) << 1) | ((p & 16) >> 1);  // swap bits 3<->4
    ln2gb[idx] = which ? ln2b[net * MIDC + K] : ln2g[net * MIDC + K];
  }
  __syncthreads();

  const int wv = tid >> 5, l = tid & 31;
  const int tile = blockIdx.x * 8 + wv;
  if (tile >= nTiles) return;
  const int base = tile * 16;
  const int el = l & 15, grp = l >> 4, col = l & 15;

  // ---- geometry / edge features -------------------------------------------
  const int eg = base + el;
  const int sA = src[eg], dA = dst[eg];
  const float dx = pos[dA * 3 + 0] - pos[sA * 3 + 0];
  const float dy = pos[dA * 3 + 1] - pos[sA * 3 + 1];
  const float dz = pos[dA * 3 + 2] - pos[sA * 3 + 2];
  const float r = sqrtf(dx * dx + dy * dy + dz * dz);
  const float rinv = 1.0f / fmaxf(r, 1e-8f);
  const float f0 = edge_w[eg * 2 + 0], f1 = edge_w[eg * 2 + 1], f2 = r;
  if (l < 16) {
    dstStash[wv * 16 + el] = dA;
    y1Stash[(wv * 16 + el) * 3 + 0] = C1F * dy * rinv;  // y
    y1Stash[(wv * 16 + el) * 3 + 1] = C1F * dz * rinv;  // z
    y1Stash[(wv * 16 + el) * 3 + 2] = C1F * dx * rinv;  // x
  }

  // ---- gather s = h0[src]: C-layout registers (for broadcasts) ------------
  float Sr[8];
#pragma unroll
  for (int v = 0; v < 8; ++v) {
    const int e = base + v + 8 * grp;
    Sr[v] = h0in[src[e] * NCH + col];
  }
  // ---- and s in A-layout (own edge's row, 8 contiguous floats, K padded) --
  v16h sa;
  {
    const float* srow = h0in + (size_t)sA * NCH + ((l & 16) ? 8 : 0);
#pragma unroll
    for (int h = 0; h < 8; ++h) sa[h] = (_Float16)srow[h];
#pragma unroll
    for (int h = 8; h < 16; ++h) sa[h] = (_Float16)0.f;
  }

  // ---- per net: radial MLP up to L2, leaving a2 in registers --------------
  v16h a2s[2];
#pragma unroll
  for (int ns = 0; ns < NNETS; ++ns) {
    const int net = (ns == 0) ? net0 : net1;
    // -- L1 (feat[3] -> 32) + LN + relu; lane does m = grp*16..+15 of edge el
    {
      const int mb = grp << 4;
      const float* W1 = rW1 + net * 3 * MIDC;
      float pre[16], ps = 0.f, ps2 = 0.f;
#pragma unroll
      for (int k = 0; k < 16; ++k) {
        const int m = mb + k;
        float t = rb1[net * MIDC + m];
        t = fmaf(f0, W1[m], t);
        t = fmaf(f1, W1[MIDC + m], t);
        t = fmaf(f2, W1[2 * MIDC + m], t);
        pre[k] = t;
        ps += t;
        ps2 += t * t;
      }
      ps = red_pair32(ps);
      ps2 = red_pair32(ps2);
      const float mu = ps * (1.f / MIDC);
      const float var = ps2 * (1.f / MIDC) - mu * mu;
      const float rs = rsqrtf(var + 1e-5f);
      v8h lo8, hi8;
#pragma unroll
      for (int k = 0; k < 8; ++k) {
        const int m = mb + k;
        lo8[k] = (_Float16)fmaxf(
            0.f, fmaf((pre[k] - mu) * rs, ln1g[net * MIDC + m],
                      ln1b[net * MIDC + m]));
        const int m2 = mb + 8 + k;
        hi8[k] = (_Float16)fmaxf(
            0.f, fmaf((pre[8 + k] - mu) * rs, ln1g[net * MIDC + m2],
                      ln1b[net * MIDC + m2]));
      }
      _Float16* bp = &hbuf[wv][el * MIDC + (grp << 3)];
      *(v8h*)(bp) = lo8;       // phys grp*8 + 0..7   (k=0..7)
      *(v8h*)(bp + 16) = hi8;  // phys grp*8 + 16..23 (k=8..15)
    }
    // -- L2 (32 -> 32): 2x WMMA (bias in C); pre-LN values round-trip
    //    through hbuf in phys-A order; LN stats = own 16 K-values + partner
    {
      const v16h a1 = load_A(&hbuf[wv][0], l);
      const float blo = rb2[net * MIDC + col];
      const float bhi = rb2[net * MIDC + 16 + col];
      v8f clo, chi;
#pragma unroll
      for (int k = 0; k < 8; ++k) { clo[k] = blo; chi[k] = bhi; }
      const v16h bl = *(const v16h*)&B2lds[((ns * 2 + 0) * 32 + l) * 16];
      const v16h bh = *(const v16h*)&B2lds[((ns * 2 + 1) * 32 + l) * 16];
      clo = wmma16(a1, bl, clo);
      chi = wmma16(a1, bh, chi);
      // store pre-LN to hbuf in phys-A layout (immediate-offset addressing)
      const int plo = (col & 7) | ((col & 8) << 1);  // phys pos of K=col
      _Float16* sb = &hbuf[wv][(8 * grp) * MIDC + plo];
#pragma unroll
      for (int v = 0; v < 8; ++v) {
        sb[v * MIDC] = (_Float16)clo[v];      // K = col
        sb[v * MIDC + 8] = (_Float16)chi[v];  // K = col+16
      }
      // reload own A-row; stats over this lane's 16 K-values + partner's 16
      const v16h x = load_A(&hbuf[wv][0], l);
      float xs[16], ps = 0.f, ps2 = 0.f;
#pragma unroll
      for (int h = 0; h < 16; ++h) {
        xs[h] = (float)x[h];
        ps += xs[h];
        ps2 += xs[h] * xs[h];
      }
      ps = red_pair32(ps);
      ps2 = red_pair32(ps2);
      const float mu = ps * (1.f / MIDC);
      const float var = ps2 * (1.f / MIDC) - mu * mu;
      const float rs = rsqrtf(var + 1e-5f);
      const float* gp = &ln2gb[(ns * 2 + 0) * 32 + (l & 16)];
      const float* bp = &ln2gb[(ns * 2 + 1) * 32 + (l & 16)];
      v16h a2;
#pragma unroll
      for (int h = 0; h < 16; ++h)
        a2[h] = (_Float16)fmaxf(0.f, fmaf((xs[h] - mu) * rs, gp[h], bp[h]));
      a2s[ns] = a2;
    }
  }

  // ---- L3 fused with einsum: acc seeded by the rb3-bias GEMM (one WMMA
  // per net: bias[e,o] = sum_i rb3[o*16+i]*s[e,i]), then 16 tau tiles with
  // zero-C WMMAs; s broadcasts shared by both nets.
  const v8f zc = {0.f, 0.f, 0.f, 0.f, 0.f, 0.f, 0.f, 0.f};
  v8f acc0 = wmma16(sa, *(const v16h*)&Bblds[(0 * 32 + l) * 16], zc);
  v8f acc1 = zc;
  if (NNETS == 2)
    acc1 = wmma16(sa, *(const v16h*)&Bblds[(1 * 32 + l) * 16], zc);
  TAU_STEP(0);  TAU_STEP(1);  TAU_STEP(2);  TAU_STEP(3);
  TAU_STEP(4);  TAU_STEP(5);  TAU_STEP(6);  TAU_STEP(7);
  TAU_STEP(8);  TAU_STEP(9);  TAU_STEP(10); TAU_STEP(11);
  TAU_STEP(12); TAU_STEP(13); TAU_STEP(14); TAU_STEP(15);

  // ---- scatter-add (segment sum) ------------------------------------------
  const int* dsb = &dstStash[wv * 16 + 8 * grp];
  const float* ysb = &y1Stash[(wv * 16 + 8 * grp) * 3];
#pragma unroll
  for (int v = 0; v < 8; ++v) {
    const int dn = dsb[v];
    atomicAdd(&out0raw[dn * NCH + col], Y0F * acc0[v]);
    if (NNETS == 2) {
      const float t = acc1[v];
      atomicAdd(&out1raw[dn * 48 + col * 3 + 0], t * ysb[v * 3 + 0]);
      atomicAdd(&out1raw[dn * 48 + col * 3 + 1], t * ysb[v * 3 + 1]);
      atomicAdd(&out1raw[dn * 48 + col * 3 + 2], t * ysb[v * 3 + 2]);
    }
  }
}

// ---------------------------------------------------------------------------
// out0 = raw*inv_deg + has_in*(h0 @ Wself^T)
__global__ void finalize0_kernel(const float* __restrict__ h0in,
                                 const float* __restrict__ out0raw,
                                 const float* __restrict__ deg,
                                 const float* __restrict__ Wself,
                                 float* __restrict__ h0out, int N) {
  int idx = blockIdx.x * blockDim.x + threadIdx.x;
  if (idx >= N * NCH) return;
  const int n = idx >> 4, o = idx & 15;
  const float dg = deg[n];
  const float inv = (dg > 0.f) ? 1.f / fmaxf(dg, 1.f) : 0.f;
  const float hi = (dg > 0.f) ? 1.f : 0.f;
  float s = 0.f;
#pragma unroll
  for (int i = 0; i < NCH; ++i)
    s = fmaf(h0in[n * NCH + i], Wself[o * NCH + i], s);
  h0out[idx] = out0raw[idx] * inv + hi * s;
}

__global__ void finalize1_kernel(const float* __restrict__ out1raw,
                                 const float* __restrict__ deg,
                                 float* __restrict__ outp, int N) {
  int idx = blockIdx.x * blockDim.x + threadIdx.x;
  if (idx >= N * 48) return;
  const int n = idx / 48;
  const float dg = deg[n];
  const float inv = (dg > 0.f) ? 1.f / fmaxf(dg, 1.f) : 0.f;
  outp[idx] = out1raw[idx] * inv;
}

// ---------------------------------------------------------------------------
extern "C" void kernel_launch(void* const* d_in, const int* in_sizes, int n_in,
                              void* d_out, int out_size, void* d_ws,
                              size_t ws_size, hipStream_t stream) {
  const float* node_feats = (const float*)d_in[0];
  const float* pos        = (const float*)d_in[1];
  const float* edge_w     = (const float*)d_in[2];
  const float* lin1_W     = (const float*)d_in[3];
  const float* lin1_b     = (const float*)d_in[4];
  const float* lin2_W     = (const float*)d_in[5];
  const float* lin2_b     = (const float*)d_in[6];
  const float* rW1        = (const float*)d_in[7];
  const float* rb1        = (const float*)d_in[8];
  const float* ln1_g      = (const float*)d_in[9];
  const float* ln1_b      = (const float*)d_in[10];
  const float* rW2        = (const float*)d_in[11];
  const float* rb2        = (const float*)d_in[12];
  const float* ln2_g      = (const float*)d_in[13];
  const float* ln2_b      = (const float*)d_in[14];
  const float* rW3        = (const float*)d_in[15];
  const float* rb3        = (const float*)d_in[16];
  const float* Wself      = (const float*)d_in[17];
  const int*   src        = (const int*)d_in[18];
  const int*   dst        = (const int*)d_in[19];

  const int N = in_sizes[1] / 3;   // 20000
  const int E = in_sizes[18];      // 320000
  const int nTiles = (E + 15) / 16;

  float* h0a   = (float*)d_ws;           // [N,16]
  float* h0b   = h0a + (size_t)N * NCH;  // [N,16]
  float* degb  = h0b + (size_t)N * NCH;  // [N]
  float* out0  = degb + N;               // [N,16]
  float* out1  = out0 + (size_t)N * NCH; // [N,48]

  float* d_out_f = (float*)d_out;

  const int TB = 256;
  // zero deg + out0 (contiguous)
  fill_kernel<<<(N + N * NCH + TB - 1) / TB, TB, 0, stream>>>(degb,
                                                              N + N * NCH, 0.f);
  embed_kernel<<<(N + TB - 1) / TB, TB, 0, stream>>>(node_feats, lin1_W, lin1_b,
                                                     lin2_W, lin2_b, h0a, N);
  deg_kernel<<<(E + TB - 1) / TB, TB, 0, stream>>>(dst, degb, E);

  const int egrid = (nTiles + 7) / 8;
  // layer 1: only net 0 (R00); layer-1 msg1 is dead in the reference
  edge_kernel<1><<<egrid, TB, 0, stream>>>(
      pos, edge_w, src, dst, rW1, rb1, ln1_g, ln1_b, rW2, rb2, ln2_g, ln2_b,
      rW3, rb3, h0a, out0, out0, 0, -1, nTiles);
  finalize0_kernel<<<(N * NCH + TB - 1) / TB, TB, 0, stream>>>(
      h0a, out0, degb, Wself, h0b, N);

  // zero out0 + out1 (contiguous)
  fill_kernel<<<(N * NCH + N * 48 + TB - 1) / TB, TB, 0, stream>>>(
      out0, N * NCH + N * 48, 0.f);

  // layer 2: nets 2 (R00) and 3 (R01)
  edge_kernel<2><<<egrid, TB, 0, stream>>>(
      pos, edge_w, src, dst, rW1, rb1, ln1_g, ln1_b, rW2, rb2, ln2_g, ln2_b,
      rW3, rb3, h0b, out0, out1, 2, 3, nTiles);
  finalize0_kernel<<<(N * NCH + TB - 1) / TB, TB, 0, stream>>>(
      h0b, out0, degb, Wself + NCH * NCH, d_out_f, N);
  finalize1_kernel<<<(N * 48 + TB - 1) / TB, TB, 0, stream>>>(
      out1, degb, d_out_f + (size_t)N * NCH, N);
}
